// MultiHeadAttention_30047591202865
// MI455X (gfx1250) — matrix operand order repaired and checked by reading, not compiled, not run on hardware
//
#include <hip/hip_runtime.h>

// ---------------------------------------------------------------- types
typedef __attribute__((ext_vector_type(16))) _Float16 v16h;
typedef __attribute__((ext_vector_type(8)))  _Float16 v8h;
typedef __attribute__((ext_vector_type(8)))  float    v8f;

#define B_  4
#define L_  2048
#define E_  1024
#define H_  16
#define DK_ 64
#define BL_ (B_ * L_)      // 8192
#define E3_ (3 * E_)       // 3072

// ---------------------------------------------------------------- WMMA
static __device__ __forceinline__ v8f wmma_f16(v16h a, v16h b, v8f c) {
  // D = A(16x32,f16) * B(32x16,f16) + C(16x16,f32)
  return __builtin_amdgcn_wmma_f32_16x16x32_f16(
      /*neg_a=*/false, a, /*neg_b=*/false, b,
      /*c_mod=*/(short)0, c, /*reuse_a=*/false, /*reuse_b=*/false);
}

// A-fragment (16 rows x 32 K) from a row-major matrix [rows, ld].
// Lane half h holds K = {8h..8h+7} U {8h+16..8h+23}  (ISA 16-bit A 16x32 table)
static __device__ __forceinline__ v16h load_frag_a(const _Float16* __restrict__ base,
                                                   int ld, int row0, int col0) {
  const int lane = threadIdx.x & 31;
  const int kb   = (lane >> 4) << 3;
  const _Float16* p = base + (size_t)(row0 + (lane & 15)) * (size_t)ld + col0 + kb;
  v8h lo = *(const v8h*)p;
  v8h hi = *(const v8h*)(p + 16);
  v16h r;
#pragma unroll
  for (int i = 0; i < 8; ++i) { r[i] = lo[i]; r[i + 8] = hi[i]; }
  return r;
}

// B-fragment (32 K x 16 cols), source given TRANSPOSED row-major: Bt[col, K].
// Lane half h holds K = {8h..8h+7} U {8h+16..8h+23} (same order as A fragment).
static __device__ __forceinline__ v16h load_frag_b(const _Float16* __restrict__ base,
                                                   int ld, int n0, int k0) {
  const int lane = threadIdx.x & 31;
  const _Float16* p = base + (size_t)(n0 + (lane & 15)) * (size_t)ld
                           + k0 + ((lane >> 4) << 3);
  v8h lo = *(const v8h*)p;
  v8h hi = *(const v8h*)(p + 16);
  v16h r;
#pragma unroll
  for (int i = 0; i < 8; ++i) { r[i] = lo[i]; r[i + 8] = hi[i]; }
  return r;
}

// ---------------------------------------------------------------- f32 -> f16 convert
__global__ void cvt_f32_f16_kernel(const float* __restrict__ in,
                                   _Float16* __restrict__ out, int n) {
  int i = (blockIdx.x * blockDim.x + threadIdx.x) * 4;
  if (i < n) {
    float4 v = *(const float4*)(in + i);
    out[i + 0] = (_Float16)v.x;
    out[i + 1] = (_Float16)v.y;
    out[i + 2] = (_Float16)v.z;
    out[i + 3] = (_Float16)v.w;
  }
}

// ---------------------------------------------------------------- QKV projection GEMM
// qkv[8192, 3072] = xh[8192,1024] @ wh[3072,1024]^T ; epilogue scatters to Q/K/Vt (f16)
// block = 256 thr = 8 waves (2 M x 4 N); wave tile 32x64; block tile 64x256
__global__ __launch_bounds__(256)
void qkv_gemm_kernel(const _Float16* __restrict__ xh,
                     const _Float16* __restrict__ wh,
                     _Float16* __restrict__ Qh,
                     _Float16* __restrict__ Kh,
                     _Float16* __restrict__ Vth) {
  const int lane = threadIdx.x & 31;
  const int wave = threadIdx.x >> 5;
  const int m0 = blockIdx.y * 64  + (wave >> 2) * 32;
  const int n0 = blockIdx.x * 256 + (wave & 3) * 64;

  v8f c[2][4] = {};
  for (int kk = 0; kk < E_; kk += 32) {
    v16h a0 = load_frag_a(xh, E_, m0,      kk);
    v16h a1 = load_frag_a(xh, E_, m0 + 16, kk);
    v16h b[4];
#pragma unroll
    for (int ni = 0; ni < 4; ++ni) b[ni] = load_frag_b(wh, E_, n0 + 16 * ni, kk);
#pragma unroll
    for (int ni = 0; ni < 4; ++ni) {
      c[0][ni] = wmma_f16(a0, b[ni], c[0][ni]);
      c[1][ni] = wmma_f16(a1, b[ni], c[1][ni]);
    }
  }

  const int half = lane >> 4, ln = lane & 15;
#pragma unroll
  for (int mi = 0; mi < 2; ++mi) {
#pragma unroll
    for (int ni = 0; ni < 4; ++ni) {
      const int gc    = n0 + 16 * ni + ln;   // 0..3071
      const int which = gc >> 10;            // 0=Q 1=K 2=V
      const int e     = gc & 1023;
      const int h     = e >> 6;
      const int d     = e & 63;
#pragma unroll
      for (int r = 0; r < 8; ++r) {
        const int gr = m0 + 16 * mi + r + 8 * half; // 0..8191
        const int b_ = gr >> 11;
        const int l  = gr & (L_ - 1);
        const float v = c[mi][ni][r];
        const size_t bh = (size_t)(b_ * H_ + h);
        if (which == 0) {
          Qh[(bh * L_ + l) * DK_ + d] = (_Float16)(v * 0.125f);  // fold 1/sqrt(64)
        } else if (which == 1) {
          Kh[(bh * L_ + l) * DK_ + d] = (_Float16)v;
        } else {
          Vth[(bh * DK_ + d) * L_ + l] = (_Float16)v;            // V transposed
        }
      }
    }
  }
}

// ---------------------------------------------------------------- flash attention
// one wave = one (b,h, 16-row query tile); 4 waves / block
__global__ __launch_bounds__(128)
void attn_kernel(const _Float16* __restrict__ Qh,
                 const _Float16* __restrict__ Kh,
                 const _Float16* __restrict__ Vth,
                 _Float16* __restrict__ ctx) {
  __shared__ _Float16 pbuf[4][16][40];   // per-wave P transpose buffer, padded
  const int lane   = threadIdx.x & 31;
  const int wlocal = threadIdx.x >> 5;
  const int gw     = blockIdx.x * 4 + wlocal;          // 0..8191
  const int qtile  = gw & (L_ / 16 - 1);               // 0..127
  const int bh     = gw >> 7;                          // 0..63
  const int q0     = qtile << 4;
  const int half   = lane >> 4, ln = lane & 15;

  const _Float16* qp = Qh  + (size_t)bh * L_ * DK_;
  const _Float16* kp = Kh  + (size_t)bh * L_ * DK_;
  const _Float16* vp = Vth + (size_t)bh * DK_ * L_;    // [DK, L]

  const v16h qa0 = load_frag_a(qp, DK_, q0, 0);
  const v16h qa1 = load_frag_a(qp, DK_, q0, 32);

  v8f o[4] = {};
  float mrun[8], lrun[8];
#pragma unroll
  for (int r = 0; r < 8; ++r) { mrun[r] = -1e30f; lrun[r] = 0.0f; }

  const int nsteps = (q0 + 16 + 31) >> 5;              // key tiles of 32
  for (int s = 0; s < nsteps; ++s) {
    const int m0 = s << 5;
    // ---- S = Q @ K^T  (two 16x16 N-tiles, chained over K-dim 64)
    v8f s0 = {}, s1 = {};
    s0 = wmma_f16(qa0, load_frag_b(kp, DK_, m0,      0),  s0);
    s0 = wmma_f16(qa1, load_frag_b(kp, DK_, m0,      32), s0);
    s1 = wmma_f16(qa0, load_frag_b(kp, DK_, m0 + 16, 0),  s1);
    s1 = wmma_f16(qa1, load_frag_b(kp, DK_, m0 + 16, 32), s1);

    // ---- causal mask + running row-max (rows live in fixed VGPR slot per half)
    float t[8];
#pragma unroll
    for (int r = 0; r < 8; ++r) {
      const int row = q0 + r + 8 * half;
      if (m0 + ln      > row) s0[r] = -1e30f;
      if (m0 + 16 + ln > row) s1[r] = -1e30f;
      t[r] = fmaxf(s0[r], s1[r]);
    }
#pragma unroll
    for (int off = 1; off <= 8; off <<= 1) {
#pragma unroll
      for (int r = 0; r < 8; ++r) t[r] = fmaxf(t[r], __shfl_xor(t[r], off, 32));
    }

    float scale[8], rs[8];
#pragma unroll
    for (int r = 0; r < 8; ++r) {
      const float mnew = fmaxf(mrun[r], t[r]);
      scale[r] = __expf(mrun[r] - mnew);
      mrun[r]  = mnew;
      s0[r] = __expf(s0[r] - mnew);
      s1[r] = __expf(s1[r] - mnew);
      rs[r] = s0[r] + s1[r];
    }
#pragma unroll
    for (int off = 1; off <= 8; off <<= 1) {
#pragma unroll
      for (int r = 0; r < 8; ++r) rs[r] += __shfl_xor(rs[r], off, 32);
    }
#pragma unroll
    for (int r = 0; r < 8; ++r) lrun[r] = lrun[r] * scale[r] + rs[r];
#pragma unroll
    for (int t4 = 0; t4 < 4; ++t4)
#pragma unroll
      for (int r = 0; r < 8; ++r) o[t4][r] *= scale[r];

    // ---- P (C-layout) -> A-fragment via per-wave LDS bounce (f16)
#pragma unroll
    for (int r = 0; r < 8; ++r) {
      pbuf[wlocal][r + 8 * half][ln]      = (_Float16)s0[r];
      pbuf[wlocal][r + 8 * half][16 + ln] = (_Float16)s1[r];
    }
    asm volatile("s_wait_dscnt 0x0" ::: "memory");  // same-wave LDS RAW
    v16h pa;
    {
      const int kb = half << 3;
      const _Float16* pr = &pbuf[wlocal][ln][kb];
      v8h plo = *(const v8h*)pr;
      v8h phi = *(const v8h*)(pr + 16);
#pragma unroll
      for (int i = 0; i < 8; ++i) { pa[i] = plo[i]; pa[i + 8] = phi[i]; }
    }

    // ---- O += P @ V  (V fragment = contiguous row of Vt)
#pragma unroll
    for (int t4 = 0; t4 < 4; ++t4) {
      v16h vb = load_frag_b(vp, L_, t4 * 16, m0);
      o[t4] = wmma_f16(pa, vb, o[t4]);
    }
  }

  // ---- epilogue: ctx[b, l, h*64+d] = O / l_sum   (f16)
  const int b_ = bh >> 4, h = bh & 15;
#pragma unroll
  for (int t4 = 0; t4 < 4; ++t4) {
#pragma unroll
    for (int r = 0; r < 8; ++r) {
      const int row = q0 + r + 8 * half;
      ctx[((size_t)(b_ * L_ + row)) * E_ + h * DK_ + t4 * 16 + ln] =
          (_Float16)(o[t4][r] / lrun[r]);
    }
  }
}

// ---------------------------------------------------------------- output projection
// out[8192,1024] (f32) = ctx[8192,1024] @ woh[1024,1024]^T
__global__ __launch_bounds__(256)
void out_gemm_kernel(const _Float16* __restrict__ ctx,
                     const _Float16* __restrict__ woh,
                     float* __restrict__ out) {
  const int lane = threadIdx.x & 31;
  const int wave = threadIdx.x >> 5;
  const int m0 = blockIdx.y * 64  + (wave >> 2) * 32;
  const int n0 = blockIdx.x * 256 + (wave & 3) * 64;

  v8f c[2][4] = {};
  for (int kk = 0; kk < E_; kk += 32) {
    v16h a0 = load_frag_a(ctx, E_, m0,      kk);
    v16h a1 = load_frag_a(ctx, E_, m0 + 16, kk);
    v16h b[4];
#pragma unroll
    for (int ni = 0; ni < 4; ++ni) b[ni] = load_frag_b(woh, E_, n0 + 16 * ni, kk);
#pragma unroll
    for (int ni = 0; ni < 4; ++ni) {
      c[0][ni] = wmma_f16(a0, b[ni], c[0][ni]);
      c[1][ni] = wmma_f16(a1, b[ni], c[1][ni]);
    }
  }

  const int half = lane >> 4, ln = lane & 15;
#pragma unroll
  for (int mi = 0; mi < 2; ++mi)
#pragma unroll
    for (int ni = 0; ni < 4; ++ni)
#pragma unroll
      for (int r = 0; r < 8; ++r) {
        const int gr = m0 + 16 * mi + r + 8 * half;
        const int gc = n0 + 16 * ni + ln;
        out[(size_t)gr * E_ + gc] = c[mi][ni][r];
      }
}

// ---------------------------------------------------------------- launch
extern "C" void kernel_launch(void* const* d_in, const int* in_sizes, int n_in,
                              void* d_out, int out_size, void* d_ws, size_t ws_size,
                              hipStream_t stream) {
  (void)in_sizes; (void)n_in; (void)out_size; (void)ws_size;
  const float* x     = (const float*)d_in[0];
  // d_in[1] = mask (unused: causality computed analytically)
  const float* w_qkv = (const float*)d_in[2];
  const float* wo    = (const float*)d_in[3];
  float* out = (float*)d_out;

  // workspace carve-up (f16 elements)
  _Float16* xh  = (_Float16*)d_ws;             // 8192*1024
  _Float16* wh  = xh  + (size_t)BL_ * E_;      // 3072*1024
  _Float16* woh = wh  + (size_t)E3_ * E_;      // 1024*1024
  _Float16* Qh  = woh + (size_t)E_  * E_;      // [B,H,L,DK]
  _Float16* Kh  = Qh  + (size_t)BL_ * E_;      // [B,H,L,DK]
  _Float16* Vth = Kh  + (size_t)BL_ * E_;      // [B,H,DK,L]
  _Float16* ctx = Vth + (size_t)BL_ * E_;      // [B,L,E]

  const int nx = BL_ * E_, nw = E3_ * E_, nwo = E_ * E_;
  cvt_f32_f16_kernel<<<nx  / 1024, 256, 0, stream>>>(x,     xh,  nx);
  cvt_f32_f16_kernel<<<nw  / 1024, 256, 0, stream>>>(w_qkv, wh,  nw);
  cvt_f32_f16_kernel<<<nwo / 1024, 256, 0, stream>>>(wo,    woh, nwo);

  qkv_gemm_kernel<<<dim3(E3_ / 256, BL_ / 64), 256, 0, stream>>>(xh, wh, Qh, Kh, Vth);
  attn_kernel<<<(B_ * H_ * (L_ / 16)) / 4, 128, 0, stream>>>(Qh, Kh, Vth, ctx);
  out_gemm_kernel<<<dim3(E_ / 256, BL_ / 64), 256, 0, stream>>>(ctx, woh, out);
}
